// GRU_21028159881856
// MI455X (gfx1250) — matrix operand order repaired and checked by reading, not compiled, not run on hardware
//
#include <hip/hip_runtime.h>
#include <hip/hip_bf16.h>

// ---------------- problem constants ----------------
#define N_NODES 10000
#define N_EDGES 160000
#define BATCH   4
#define T_STEPS 16
#define HID     64
#define CIN     65          // 1 + HID
#define M_TOTAL (BATCH * N_NODES)   // 40000 rows
#define M_TILES (M_TOTAL / 16)      // 2500 (16 | 10000, so tiles never straddle a batch)
#define TILES_PER_BATCH (N_NODES / 16) // 625

typedef __attribute__((ext_vector_type(16))) _Float16 v16h;
typedef __attribute__((ext_vector_type(8)))  float    v8f;

union Frag16 { v16h v; _Float16 h[16]; };

__device__ __forceinline__ float sigmoidf_(float x) {
  return 1.0f / (1.0f + __expf(-x));
}

// ---------------- generic zero ----------------
__global__ void zero_f(float* p, int count) {
  int i = blockIdx.x * blockDim.x + threadIdx.x;
  int stride = gridDim.x * blockDim.x;
  for (; i < count; i += stride) p[i] = 0.0f;
}

// ---------------- weight prep (once per call) ----------------
// w16 layout: [gate(3)][mat(3)][n(64)][k(64)] f16, where k maps input channel c=k+1.
//   mat 0: W[0,0]+W[1,0]   mat 1: W[0,1]   mat 2: W[1,1]
// w0 layout: [gate][mat][n] f32 (channel 0 column). bias: [gate][n] f32.
__global__ void prep_weights(const float* Wz, const float* bz,
                             const float* Wr, const float* br,
                             const float* Wh, const float* bh,
                             _Float16* w16, float* w0, float* bias) {
  const float* Wg[3] = {Wz, Wr, Wh};
  const float* bg[3] = {bz, br, bh};
  int stride = gridDim.x * blockDim.x;
  int tid = blockIdx.x * blockDim.x + threadIdx.x;
  // f16 part
  for (int i = tid; i < 3 * 3 * 64 * 64; i += stride) {
    int k    = i & 63;
    int n    = (i >> 6) & 63;
    int mat  = (i >> 12) % 3;
    int gate = (i >> 12) / 3;
    int c = k + 1;
    const float* W = Wg[gate];
    // W flat: ((d*2+o)*65 + c)*64 + n ; blocks: (0,0)=0 (0,1)=1 (1,0)=2 (1,1)=3
    float v;
    if (mat == 0)      v = W[(0 * CIN + c) * 64 + n] + W[(2 * CIN + c) * 64 + n];
    else if (mat == 1) v = W[(1 * CIN + c) * 64 + n];
    else               v = W[(3 * CIN + c) * 64 + n];
    w16[i] = (_Float16)v;
  }
  // channel-0 columns (fp32, exact)
  for (int i = tid; i < 3 * 3 * 64; i += stride) {
    int n    = i & 63;
    int mat  = (i >> 6) % 3;
    int gate = (i >> 6) / 3;
    const float* W = Wg[gate];
    float v;
    if (mat == 0)      v = W[(0 * CIN + 0) * 64 + n] + W[(2 * CIN + 0) * 64 + n];
    else if (mat == 1) v = W[(1 * CIN + 0) * 64 + n];
    else               v = W[(3 * CIN + 0) * 64 + n];
    w0[i] = v;
  }
  for (int i = tid; i < 3 * 64; i += stride) bias[i] = bg[i >> 6][i & 63];
}

// ---------------- degrees & edge coefficients ----------------
__global__ void scatter_deg(const int* row, const int* col, const float* ew,
                            float* deg_out, float* deg_in) {
  int e = blockIdx.x * blockDim.x + threadIdx.x;
  if (e >= N_EDGES) return;
  atomicAdd(deg_out + row[e], ew[e]);
  atomicAdd(deg_in  + col[e], ew[e]);
}

__global__ void edge_coef(const int* row, const int* col, const float* ew,
                          const float* deg_out, const float* deg_in,
                          float* coef_f, float* coef_b) {
  int e = blockIdx.x * blockDim.x + threadIdx.x;
  if (e >= N_EDGES) return;
  float dout = deg_out[row[e]];
  float din  = deg_in[col[e]];
  coef_f[e] = ew[e] * (dout > 0.0f ? 1.0f / dout : 0.0f);
  coef_b[e] = ew[e] * (din  > 0.0f ? 1.0f / din  : 0.0f);
}

// ---------------- edge scatter:  agg_f[:,row] += cf * xh[:,col],  agg_b[:,col] += cb * xh[:,row]
// xh channel 0 = x_t, channels 1..64 = Hsrc (h for z/r, r*h for c)
__global__ void scatter_edges(const int* rowIdx, const int* colIdx,
                              const float* coef_f, const float* coef_b,
                              const float* x_dis, int t,
                              const float* Hsrc,
                              float* aggf, float* aggb) {
  int tid = blockIdx.x * blockDim.x + threadIdx.x;
  if (tid >= N_EDGES * BATCH) return;
  int e = tid >> 2;
  int b = tid & 3;
  int r = rowIdx[e], c = colIdx[e];
  float cf = coef_f[e], cb = coef_b[e];
  const float* xt = x_dis + (size_t)(b * T_STEPS + t) * N_NODES;
  // channel 0
  atomicAdd(aggf + ((size_t)(b * N_NODES + r)) * CIN, cf * xt[c]);
  atomicAdd(aggb + ((size_t)(b * N_NODES + c)) * CIN, cb * xt[r]);
  // channels 1..64
  const float* hc = Hsrc + ((size_t)(b * N_NODES + c)) * HID;
  const float* hr = Hsrc + ((size_t)(b * N_NODES + r)) * HID;
  float* af = aggf + ((size_t)(b * N_NODES + r)) * CIN + 1;
  float* ab = aggb + ((size_t)(b * N_NODES + c)) * CIN + 1;
#pragma unroll 4
  for (int k = 0; k < HID; ++k) {
    atomicAdd(af + k, cf * hc[k]);
    atomicAdd(ab + k, cb * hr[k]);
  }
}

// ---------------- WMMA gate GEMM ----------------
// One block = one 16-row M-tile, 4 waves = 4 column tiles (n0 = wave*16).
// A (16x64 f16) staged in LDS (padded stride 72). B fragments loaded from
// pre-transposed f16 weights [n][k] in global (L2 resident).
template <bool IS_ZR>
__global__ __launch_bounds__(128)
void gates_kernel(const float* Asrc,      // h (z/r) or r*h (c) : [B,N,64]
                  const float* aggf, const float* aggb,   // [B,N,65]
                  const float* x_dis, int t,
                  const _Float16* w16, const float* w0, const float* bias,
                  const float* h_state, float* z_buf, float* rh_buf, float* h_out) {
  __shared__ _Float16 sA[3][16][72];
  __shared__ float    sC0[3][16];

  const int b      = blockIdx.x / TILES_PER_BATCH;
  const int n_base = (blockIdx.x % TILES_PER_BATCH) * 16;
  const int tid    = threadIdx.x;

  // ---- stage A tiles (fp32 -> f16) ----
  for (int idx = tid; idx < 3 * 16 * HID; idx += 128) {
    int tile = idx >> 10;            // /1024
    int rem  = idx & 1023;
    int rowL = rem >> 6;
    int k    = rem & 63;
    size_t node = (size_t)(b * N_NODES + n_base + rowL);
    float v;
    if (tile == 0)      v = Asrc[node * HID + k];
    else if (tile == 1) v = aggf[node * CIN + 1 + k];
    else                v = aggb[node * CIN + 1 + k];
    sA[tile][rowL][k] = (_Float16)v;
  }
  if (tid < 48) {
    int tile = tid >> 4;
    int rowL = tid & 15;
    size_t node = (size_t)(b * N_NODES + n_base + rowL);
    float v;
    if (tile == 0)      v = x_dis[(size_t)(b * T_STEPS + t) * N_NODES + n_base + rowL];
    else if (tile == 1) v = aggf[node * CIN];
    else                v = aggb[node * CIN];
    sC0[tile][rowL] = v;
  }
  __syncthreads();

  const int lane = tid & 31;
  const int wave = tid >> 5;
  const int n0   = wave * 16;
  const int m_l  = lane & 15;
  const int g    = lane >> 4;
  const int n_l  = lane & 15;

  // ---- build A fragments (ISA 7.12.2 16-bit A 16x32 layout) ----
  Frag16 fa[3][2];
#pragma unroll
  for (int tt = 0; tt < 3; ++tt) {
#pragma unroll
    for (int kc = 0; kc < 2; ++kc) {
#pragma unroll
      for (int v = 0; v < 8; ++v) {
        int kb = ((v < 4) ? (2 * v) : (16 + 2 * (v - 4))) + 8 * g + 32 * kc;
        fa[tt][kc].h[2 * v]     = sA[tt][m_l][kb];
        fa[tt][kc].h[2 * v + 1] = sA[tt][m_l][kb + 1];
      }
    }
  }

  const int gate_lo = IS_ZR ? 0 : 2;
  const int gate_hi = IS_ZR ? 1 : 2;
  for (int gate = gate_lo; gate <= gate_hi; ++gate) {
    v8f acc = {0.f, 0.f, 0.f, 0.f, 0.f, 0.f, 0.f, 0.f};
#pragma unroll
    for (int mat = 0; mat < 3; ++mat) {
      const _Float16* Wt = w16 + ((size_t)((gate * 3 + mat) * 64 + (n0 + n_l))) * 64;
#pragma unroll
      for (int kc = 0; kc < 2; ++kc) {
        Frag16 fb;
#pragma unroll
        for (int v = 0; v < 8; ++v) {
          int k = ((v < 4) ? (2 * v) : (16 + 2 * (v - 4))) + 8 * g + 32 * kc;   // B 32x16 layout: same K order as A (lanes 0-15 K 0-7,16-23; lanes 16-31 K 8-15,24-31)
          fb.h[2 * v]     = Wt[k];
          fb.h[2 * v + 1] = Wt[k + 1];
        }
        acc = __builtin_amdgcn_wmma_f32_16x16x32_f16(
            false, fa[mat][kc].v, false, fb.v, (short)0, acc, false, false);
      }
    }
    // ---- epilogue: channel-0 rank-1 (exact fp32), bias, activation, update ----
    const int n = n0 + n_l;
    float b0  = bias[gate * 64 + n];
    float w0s = w0[(gate * 3 + 0) * 64 + n];
    float w0f = w0[(gate * 3 + 1) * 64 + n];
    float w0b = w0[(gate * 3 + 2) * 64 + n];
#pragma unroll
    for (int r = 0; r < 8; ++r) {
      int m_local = r + 8 * g;                // C/D layout: VGPR r -> M = r + 8*(lane>>4)
      float pre = acc[r]
                + sC0[0][m_local] * w0s
                + sC0[1][m_local] * w0f
                + sC0[2][m_local] * w0b
                + b0;
      size_t idx = ((size_t)(b * N_NODES) + n_base + m_local) * HID + n;
      if (gate == 0) {
        z_buf[idx] = sigmoidf_(pre);
      } else if (gate == 1) {
        float rv = sigmoidf_(pre);
        rh_buf[idx] = rv * h_state[idx];
      } else {
        float cv = tanhf(pre);
        float zv = z_buf[idx];
        float hv = h_state[idx];
        h_out[idx] = zv * hv + (1.0f - zv) * cv;
      }
    }
  }
}

// ---------------- readout: out[b] = relu(h[b,0,:]@Wro1 + b1) @ Wro2 + b2 ----------------
__global__ void readout_kernel(const float* h, const float* Wro1, const float* bro1,
                               const float* Wro2, const float* bro2, float* out) {
  __shared__ float red[256];
  int tid = threadIdx.x;           // 256 threads: 4 groups of 64
  int b = tid >> 6;
  int j = tid & 63;
  const float* hb = h + (size_t)b * N_NODES * HID;   // node 0
  float acc = bro1[j];
  for (int k = 0; k < HID; ++k) acc += hb[k] * Wro1[k * HID + j];
  acc = fmaxf(acc, 0.0f) * Wro2[j];
  red[tid] = acc;
  __syncthreads();
  for (int s = 32; s > 0; s >>= 1) {
    if (j < s) red[tid] += red[tid + s];
    __syncthreads();
  }
  if (j == 0) out[b] = red[tid] + bro2[0];
}

// ---------------- host side ----------------
extern "C" void kernel_launch(void* const* d_in, const int* in_sizes, int n_in,
                              void* d_out, int out_size, void* d_ws, size_t ws_size,
                              hipStream_t stream) {
  const float* x_dis = (const float*)d_in[0];
  const int*   eidx  = (const int*)d_in[1];
  const float* ew    = (const float*)d_in[2];
  const float* W_z   = (const float*)d_in[3];
  const float* b_z   = (const float*)d_in[4];
  const float* W_r   = (const float*)d_in[5];
  const float* b_r   = (const float*)d_in[6];
  const float* W_h   = (const float*)d_in[7];
  const float* b_h   = (const float*)d_in[8];
  const float* W_ro1 = (const float*)d_in[9];
  const float* b_ro1 = (const float*)d_in[10];
  const float* W_ro2 = (const float*)d_in[11];
  const float* b_ro2 = (const float*)d_in[12];
  float* out = (float*)d_out;

  const int* row = eidx;
  const int* col = eidx + N_EDGES;

  // ---- carve workspace ----
  char* wsp = (char*)d_ws;
  auto carve = [&](size_t bytes) -> void* {
    void* p = (void*)wsp;
    wsp += (bytes + 255) & ~(size_t)255;
    return p;
  };
  float*    h        = (float*)carve((size_t)M_TOTAL * HID * 4);
  float*    z_buf    = (float*)carve((size_t)M_TOTAL * HID * 4);
  float*    rh_buf   = (float*)carve((size_t)M_TOTAL * HID * 4);
  float*    aggf     = (float*)carve((size_t)M_TOTAL * CIN * 4);   // aggb directly follows
  float*    aggb     = (float*)carve((size_t)M_TOTAL * CIN * 4);
  float*    deg_out  = (float*)carve((size_t)N_NODES * 4);          // deg_in follows contiguously? use carve
  float*    deg_in   = (float*)carve((size_t)N_NODES * 4);
  float*    coef_f   = (float*)carve((size_t)N_EDGES * 4);
  float*    coef_b   = (float*)carve((size_t)N_EDGES * 4);
  _Float16* w16      = (_Float16*)carve((size_t)3 * 3 * 64 * 64 * 2);
  float*    w0       = (float*)carve((size_t)3 * 3 * 64 * 4);
  float*    biasb    = (float*)carve((size_t)3 * 64 * 4);
  (void)ws_size; (void)in_sizes; (void)n_in; (void)out_size;

  // ---- one-time per call ----
  prep_weights<<<64, 256, 0, stream>>>(W_z, b_z, W_r, b_r, W_h, b_h, w16, w0, biasb);
  zero_f<<<1024, 256, 0, stream>>>(h, M_TOTAL * HID);
  zero_f<<<64, 256, 0, stream>>>(deg_out, N_NODES);
  zero_f<<<64, 256, 0, stream>>>(deg_in, N_NODES);
  scatter_deg<<<(N_EDGES + 255) / 256, 256, 0, stream>>>(row, col, ew, deg_out, deg_in);
  edge_coef<<<(N_EDGES + 255) / 256, 256, 0, stream>>>(row, col, ew, deg_out, deg_in, coef_f, coef_b);

  const int agg_count = M_TOTAL * CIN * 2;          // aggf + aggb (contiguous after 256B-align: both sizes are 256B multiples)
  const int scat_blocks = (N_EDGES * BATCH + 255) / 256;

  for (int t = 0; t < T_STEPS; ++t) {
    // z, r gates: xh = [x_t, h]
    zero_f<<<2048, 256, 0, stream>>>(aggf, agg_count);
    scatter_edges<<<scat_blocks, 256, 0, stream>>>(row, col, coef_f, coef_b,
                                                   x_dis, t, h, aggf, aggb);
    gates_kernel<true><<<M_TILES, 128, 0, stream>>>(h, aggf, aggb, x_dis, t,
                                                    w16, w0, biasb,
                                                    h, z_buf, rh_buf, nullptr);
    // candidate gate: xrh = [x_t, r*h]
    zero_f<<<2048, 256, 0, stream>>>(aggf, agg_count);
    scatter_edges<<<scat_blocks, 256, 0, stream>>>(row, col, coef_f, coef_b,
                                                   x_dis, t, rh_buf, aggf, aggb);
    gates_kernel<false><<<M_TILES, 128, 0, stream>>>(rh_buf, aggf, aggb, x_dis, t,
                                                     w16, w0, biasb,
                                                     h, z_buf, rh_buf, h);
  }

  readout_kernel<<<1, 256, 0, stream>>>(h, W_ro1, b_ro1, W_ro2, b_ro2, out);
}
